// Model_40080634806464
// MI455X (gfx1250) — matrix operand order repaired and checked by reading, not compiled, not run on hardware
//
#include <hip/hip_runtime.h>
#include <hip/hip_bf16.h>

typedef __attribute__((ext_vector_type(16))) __bf16 v16bf;
typedef __attribute__((ext_vector_type(8)))  __bf16 v8bf;
typedef __attribute__((ext_vector_type(8)))  float  v8f;

#define N_BATCH 64
#define C_IN 3
#define T_FRAMES 300
#define V_J 25
#define WINDOW 5
#define U_DIM 125           // WINDOW*V
#define NSCALE 6
#define EMBED 96
#define K3 480              // EMBED*WINDOW
#define TT 4                // t-tile

// ---- workspace layout (bytes) ----
#define WS_ABF    0                      // bf16 [6][128][128]  (A_SCALES, padded)
#define WS_WM     196608                 // bf16 [96][32]       (mlp_w, k padded 18->32)
#define WS_WOUT   202752                 // bf16 [96][480]      (out_w flat = o*480+c*5+w)
#define WS_CST    294912                 // f32  [4][96]        scale1,shift1,scale2,shift2

// ---- LDS layout (bytes) ----
#define L_XWT   0                        // bf16 [12][128]      (c*4+t rows, u padded to 128)
#define L_AGG   3072                     // bf16 [128][4][32]   (v' rows, k padded 18->32)
#define L_H3    35840                    // bf16 [112][480]     (rows v*4+t, v padded 25->28)
#define L_CST   143360                   // f32  [384]
#define LDS_BYTES 144896

__device__ __forceinline__ __bf16 f2bf(float f) {
    union { float f; unsigned u; } v; v.f = f;
    unsigned r = v.u + 0x7FFFu + ((v.u >> 16) & 1u);   // RNE
    unsigned short h = (unsigned short)(r >> 16);
    return __builtin_bit_cast(__bf16, h);
}

// Build a 16-element bf16 fragment from two contiguous 16B runs.
__device__ __forceinline__ v16bf make_frag(const __bf16* p0, const __bf16* p1) {
    v8bf lo = *(const v8bf*)p0;
    v8bf hi = *(const v8bf*)p1;
    v16bf f;
#pragma unroll
    for (int i = 0; i < 8; i++) { f[i] = lo[i]; f[i + 8] = hi[i]; }
    return f;
}

// A-operand (16x32 bf16, row-major [row][k...]): lane half g -> contiguous run kb+g*16 .. +15
__device__ __forceinline__ v16bf frag_a(const __bf16* row, int kb, int g) {
    return make_frag(row + kb + g * 16, row + kb + g * 16 + 8);
}
// B-operand (32x16 bf16, stored per-column [col][k...]): lane half g -> contiguous run kb+g*16 .. +15
__device__ __forceinline__ v16bf frag_b(const __bf16* colrow, int kb, int g) {
    return make_frag(colrow + kb + g * 16, colrow + kb + g * 16 + 8);
}

__device__ __constant__ int EDGE0[24] = {1,2,3,4,5,6,7,8,9,10,11,12,13,14,15,16,17,18,19,20,22,23,24,25};
__device__ __constant__ int EDGE1[24] = {2,21,21,3,21,5,6,7,21,9,10,11,1,13,14,15,1,17,18,19,23,8,25,12};

// ============ kernel 0: build A_SCALES + bf16 weights + folded BN constants ============
__global__ void precompute_kernel(
    const float* __restrict__ mlp_w, const float* __restrict__ mlp_b,
    const float* __restrict__ g1, const float* __restrict__ b1,
    const float* __restrict__ m1, const float* __restrict__ v1,
    const float* __restrict__ out_w, const float* __restrict__ out_b,
    const float* __restrict__ g2, const float* __restrict__ b2,
    const float* __restrict__ m2, const float* __restrict__ v2,
    __bf16* __restrict__ A_bf, __bf16* __restrict__ Wm_bf,
    __bf16* __restrict__ Wout_bf, float* __restrict__ cst)
{
    __shared__ unsigned char A25[25][25];
    __shared__ unsigned char Bm[125][125];   // bool(A_large + I)
    __shared__ unsigned char Pa[125][125];   // reachability ping
    __shared__ unsigned char Pb[125][125];   // reachability pong
    __shared__ float dinv[125];
    const int tid = threadIdx.x, NT = blockDim.x;

    for (int i = tid; i < 625; i += NT) A25[i / 25][i % 25] = 0;
    __syncthreads();
    if (tid < 24) {
        A25[EDGE0[tid] - 1][EDGE1[tid] - 1] = 1;
        A25[EDGE1[tid] - 1][EDGE0[tid] - 1] = 1;
    }
    __syncthreads();
    for (int idx = tid; idx < 125 * 125; idx += NT) {
        int i = idx / 125, j = idx % 125;
        Bm[i][j] = (A25[i % 25][j % 25] | ((i % 25) == (j % 25))) ? 1 : 0;
        Pa[i][j] = (i == j) ? 1 : 0;                 // P_0 = I
    }
    // k = 0: normalized(I) = I
    for (int idx = tid; idx < 128 * 128; idx += NT) {
        int i = idx / 128, j = idx % 128;
        A_bf[idx] = f2bf((i < 125 && i == j) ? 1.0f : 0.0f);
    }
    unsigned char (*Pp)[125] = Pa;
    unsigned char (*Pc)[125] = Pb;
    for (int k = 1; k < NSCALE; k++) {
        __syncthreads();
        for (int idx = tid; idx < 125 * 125; idx += NT) {   // boolean matmul: Pc = Pp * Bm
            int i = idx / 125, j = idx % 125;
            unsigned char r = 0;
            for (int m = 0; m < 125; m++) r |= (unsigned char)(Pp[i][m] & Bm[m][j]);
            Pc[i][j] = r;
        }
        __syncthreads();
        for (int i = tid; i < 125; i += NT) {               // row degrees of g = (Pc & !Pp) | I
            int s = 0;
            for (int j = 0; j < 125; j++)
                s += (i == j) ? 1 : (int)(Pc[i][j] & (unsigned char)(~Pp[i][j]) & 1);
            dinv[i] = (s > 0) ? (float)(1.0 / sqrt((double)s)) : 0.0f;
        }
        __syncthreads();
        for (int idx = tid; idx < 128 * 128; idx += NT) {   // D g D, bf16, padded to 128x128
            int i = idx / 128, j = idx % 128;
            float val = 0.0f;
            if (i < 125 && j < 125) {
                int gij = (i == j) ? 1 : (int)(Pc[i][j] & (unsigned char)(~Pp[i][j]) & 1);
                if (gij) val = (float)((double)dinv[i] * (double)dinv[j]);
            }
            A_bf[k * 16384 + idx] = f2bf(val);
        }
        unsigned char (*t)[125] = Pp; Pp = Pc; Pc = t;       // ping-pong
    }
    // weights + folded constants
    for (int idx = tid; idx < EMBED * 32; idx += NT) {
        int o = idx / 32, k = idx % 32;
        Wm_bf[idx] = f2bf(k < 18 ? mlp_w[o * 18 + k] : 0.0f);
    }
    for (int idx = tid; idx < EMBED * K3; idx += NT) Wout_bf[idx] = f2bf(out_w[idx]);
    for (int o = tid; o < EMBED; o += NT) {
        float inv1 = g1[o] * rsqrtf(v1[o] + 1e-5f);
        cst[o]        = inv1;
        cst[96 + o]   = (mlp_b[o] - m1[o]) * inv1 + b1[o];
        float inv2 = g2[o] * rsqrtf(v2[o] + 1e-5f);
        cst[192 + o]  = inv2;
        cst[288 + o]  = (out_b[o] - m2[o]) * inv2 + b2[o];
    }
}

// ============ kernel 1: fully fused graph-conv pipeline (per n, 4-frame tile) ============
__global__ __launch_bounds__(256) void fused_kernel(
    const float* __restrict__ x,
    const __bf16* __restrict__ A_bf, const __bf16* __restrict__ Wm_bf,
    const __bf16* __restrict__ Wout_bf, const float* __restrict__ cstg,
    float* __restrict__ out)
{
    extern __shared__ __align__(16) char smem[];
    __bf16* xwt  = (__bf16*)(smem + L_XWT);   // [c*4+t][u128]
    __bf16* agg2 = (__bf16*)(smem + L_AGG);   // [(v'*4+t)][k32]
    __bf16* h3   = (__bf16*)(smem + L_H3);    // [(v*4+t)][k3=480]
    float*  cst  = (float*)(smem + L_CST);

    const int tid  = threadIdx.x;
    const int lane = tid & 31, wv = tid >> 5;
    const int g = (lane >> 4) & 1, m = lane & 15;
    const int n  = blockIdx.y;
    const int t0 = blockIdx.x * TT;

    // ---- phase 0: stage constants, zero pads, build bf16 window tile ----
    for (int i = tid; i < 384; i += 256) cst[i] = cstg[i];
    for (int i = tid; i < (128 * TT * 32) / 2; i += 256) ((unsigned*)agg2)[i] = 0u;
    for (int i = tid; i < (12 * K3) / 2; i += 256) ((unsigned*)(h3 + 100 * K3))[i] = 0u; // v=25..27 pad rows
    for (int idx = tid; idx < 12 * 128; idx += 256) {
        int row = idx / 128, u = idx % 128;
        float val = 0.0f;
        if (u < U_DIM) {
            int c = row / TT, t = row % TT, w = u / V_J, v = u % V_J;
            int tg = t0 + t + w - 2;
            if (tg >= 0 && tg < T_FRAMES) val = x[((n * C_IN + c) * T_FRAMES + tg) * V_J + v];
        }
        xwt[idx] = f2bf(val);
    }
    __syncthreads();

    // ---- phase 1: agg[(s,c)][v'][t] = A_s @ xw   (M=v', N=(c,t), K=u 125->128) ----
    for (int tile = wv; tile < NSCALE * 8; tile += 8) {
        int s = tile / 8, vt = tile % 8;
        v8f acc;
#pragma unroll
        for (int i = 0; i < 8; i++) acc[i] = 0.0f;
        const __bf16* Arow = A_bf + (size_t)(s * 128 + vt * 16 + m) * 128;
        int c = m >> 2; if (c > 2) c = 2;          // pad cols clamp (result discarded)
        const __bf16* Brow = xwt + (c * TT + (m & 3)) * 128;
#pragma unroll
        for (int ks = 0; ks < 4; ks++) {
            v16bf a = frag_a(Arow, ks * 32, g);
            v16bf b = frag_b(Brow, ks * 32, g);
            acc = __builtin_amdgcn_wmma_f32_16x16x32_bf16(false, a, false, b,
                                                          (short)0, acc, false, false);
        }
        if (m < 12) {                               // real columns only
            int cc = m >> 2, tl = m & 3, kk = s * C_IN + cc;
#pragma unroll
            for (int r = 0; r < 8; r++) {
                int vp = vt * 16 + g * 8 + r;
                if (vp < U_DIM) agg2[(vp * TT + tl) * 32 + kk] = f2bf(acc[r]);
            }
        }
    }
    __syncthreads();

    // ---- phase 2: h = relu(bn(mlp_w @ agg))  (M=o, N=(v',t), K=18->32), scatter to h3 ----
    for (int tile = wv; tile < 32 * 6; tile += 8) {
        int p = tile / 6, ot = tile % 6;
        v8f acc;
#pragma unroll
        for (int i = 0; i < 8; i++) acc[i] = 0.0f;
        const __bf16* Arow = Wm_bf + (ot * 16 + m) * 32;
        int vp = p * TT + (m >> 2), tl = m & 3;    // v' = 0..127 (pad rows are zero)
        const __bf16* Brow = agg2 + (vp * TT + tl) * 32;
        v16bf a = frag_a(Arow, 0, g);
        v16bf b = frag_b(Brow, 0, g);
        acc = __builtin_amdgcn_wmma_f32_16x16x32_bf16(false, a, false, b,
                                                      (short)0, acc, false, false);
        if (vp < U_DIM) {
            int w = vp / V_J, v = vp % V_J;
#pragma unroll
            for (int r = 0; r < 8; r++) {
                int o = ot * 16 + g * 8 + r;
                float hv = acc[r] * cst[o] + cst[96 + o];
                hv = hv > 0.0f ? hv : 0.0f;        // ReLU
                h3[(v * TT + tl) * K3 + o * WINDOW + w] = f2bf(hv);
            }
        }
    }
    __syncthreads();

    // ---- phase 3: out = bn(Wout @ h3)   (M=o, N=(v,t), K=480 = 15x32) ----
    for (int tile = wv; tile < 6 * 7; tile += 8) {
        int ot = tile / 7, vt = tile % 7;
        v8f acc;
#pragma unroll
        for (int i = 0; i < 8; i++) acc[i] = 0.0f;
        const __bf16* Arow = Wout_bf + (size_t)(ot * 16 + m) * K3;
        int v = vt * TT + (m >> 2), tl = m & 3;    // v up to 27 (pad rows zero)
        const __bf16* Brow = h3 + (v * TT + tl) * K3;
#pragma unroll
        for (int ks = 0; ks < 15; ks++) {
            v16bf a = frag_a(Arow, ks * 32, g);
            v16bf b = frag_b(Brow, ks * 32, g);
            acc = __builtin_amdgcn_wmma_f32_16x16x32_bf16(false, a, false, b,
                                                          (short)0, acc, false, false);
        }
        int t = t0 + tl;
        if (v < V_J && t < T_FRAMES) {
#pragma unroll
            for (int r = 0; r < 8; r++) {
                int o = ot * 16 + g * 8 + r;
                out[((size_t)(n * EMBED + o) * T_FRAMES + t) * V_J + v] =
                    acc[r] * cst[192 + o] + cst[288 + o];
            }
        }
    }
}

extern "C" void kernel_launch(void* const* d_in, const int* in_sizes, int n_in,
                              void* d_out, int out_size, void* d_ws, size_t ws_size,
                              hipStream_t stream) {
    const float* x      = (const float*)d_in[0];
    const float* mlp_w  = (const float*)d_in[1];
    const float* mlp_b  = (const float*)d_in[2];
    const float* g1     = (const float*)d_in[3];
    const float* b1     = (const float*)d_in[4];
    const float* m1     = (const float*)d_in[5];
    const float* v1     = (const float*)d_in[6];
    const float* out_w  = (const float*)d_in[7];
    const float* out_b  = (const float*)d_in[8];
    const float* g2     = (const float*)d_in[9];
    const float* b2     = (const float*)d_in[10];
    const float* m2     = (const float*)d_in[11];
    const float* v2     = (const float*)d_in[12];

    char* ws = (char*)d_ws;
    __bf16* A_bf    = (__bf16*)(ws + WS_ABF);
    __bf16* Wm_bf   = (__bf16*)(ws + WS_WM);
    __bf16* Wout_bf = (__bf16*)(ws + WS_WOUT);
    float*  cst     = (float*)(ws + WS_CST);

    precompute_kernel<<<1, 128, 0, stream>>>(mlp_w, mlp_b, g1, b1, m1, v1,
                                             out_w, out_b, g2, b2, m2, v2,
                                             A_bf, Wm_bf, Wout_bf, cst);

    dim3 grid(T_FRAMES / TT, N_BATCH);   // 75 x 64
    fused_kernel<<<grid, 256, LDS_BYTES, stream>>>(x, A_bf, Wm_bf, Wout_bf, cst,
                                                   (float*)d_out);
}
